// CausalAttention_66185446031979
// MI455X (gfx1250) — matrix operand order repaired and checked by reading, not compiled, not run on hardware
//
#include <hip/hip_runtime.h>
#include <hip/hip_bf16.h>

#define USE_ASYNC_LDS 1

#if __has_builtin(__builtin_amdgcn_tensor_load_to_lds)
  #define HAVE_TDM 1
  #if __has_include(<hip/amd_detail/amd_gfx1250_TDM.h>)
    #define TDM_6ARG 1
  #endif
#else
  #define HAVE_TDM 0
#endif

// ---------------------------------------------------------------------------
// Types matching gfx1250 WMMA fragment shapes (wave32):
//   A/B: 16 bf16 elements per lane (8 VGPRs), C/D: 8 floats per lane.
// ---------------------------------------------------------------------------
typedef __bf16 v16bf __attribute__((ext_vector_type(16)));
typedef __bf16 v8bf  __attribute__((ext_vector_type(8)));
typedef float  v8f   __attribute__((ext_vector_type(8)));
typedef unsigned int v4u __attribute__((ext_vector_type(4)));
typedef int v8i __attribute__((ext_vector_type(8)));
typedef int v4i __attribute__((ext_vector_type(4)));

#define SEQ_L   2048
#define DIM_D   2048
#define HEADS   16
#define KVH     4
#define HDK     128
#define LN10000 9.210340371976184f
#define QSCALE  0.08838834764831845f   // 1/sqrt(128)

// ---------------------------------------------------------------------------
// 16-byte global -> LDS copy, ASYNCcnt-tracked (fallback staging path).
// ---------------------------------------------------------------------------
__device__ __forceinline__ void cp16_lds(__bf16* lds_dst, const __bf16* g) {
#if USE_ASYNC_LDS
  unsigned lds_off = (unsigned)(size_t)lds_dst;   // LDS aperture: addr[31:0]
  asm volatile("global_load_async_to_lds_b128 %0, %1, off"
               :: "v"(lds_off), "v"(g) : "memory");
#else
  *reinterpret_cast<v8bf*>(lds_dst) = *reinterpret_cast<const v8bf*>(g);
#endif
}

#if HAVE_TDM
// ---------------------------------------------------------------------------
// TDM 2-D tile load: one wave issues a whole-tile DMA (global -> LDS).
// D# per CDNA5 ISA §8.3/8.4: group0 = {count/type, lds_addr, global_addr},
// group1 = {data_size, tensor dims, tile dims, dim0 stride}. 2-D tensor ->
// groups 2/3 zero (VADDR2/3 semantics: unused dims).
// ---------------------------------------------------------------------------
__device__ __forceinline__ void tdm_load_2d(unsigned lds_off, const void* gp,
                                            unsigned tile_d0, unsigned tile_d1,
                                            unsigned tensor_d0,
                                            unsigned tensor_d1,
                                            unsigned stride0_elems) {
  unsigned long long ga = (unsigned long long)gp;
  v4u g0;
  g0[0] = 1u;                                   // count=1, user mode, no gather
  g0[1] = lds_off;                              // lds_addr[31:0]
  g0[2] = (unsigned)(ga & 0xffffffffu);         // global_addr[31:0]
  g0[3] = (unsigned)((ga >> 32) & 0x01ffffffu)  // global_addr[56:32]
        | (2u << 30);                           // type = 2 ("image")
  v8i g1;
  g1[0] = (int)(1u << 16);                      // data_size=1 -> 2 bytes
  g1[1] = (int)((tensor_d0 & 0xffffu) << 16);   // tensor_dim0[15:0] @ b63:48
  g1[2] = (int)(((tensor_d0 >> 16) & 0xffffu) | ((tensor_d1 & 0xffffu) << 16));
  g1[3] = (int)(((tensor_d1 >> 16) & 0xffffu) | ((tile_d0 & 0xffffu) << 16));
  g1[4] = (int)(tile_d1 & 0xffffu);             // tile_dim1; tile_dim2=0
  g1[5] = (int)stride0_elems;                   // tensor_dim0_stride[31:0]
  g1[6] = 0;                                    // stride0[47:32], dim1_stride
  g1[7] = 0;
  v4i z4 = {0, 0, 0, 0};
#if defined(TDM_6ARG)
  v8i z8 = {0, 0, 0, 0, 0, 0, 0, 0};
  __builtin_amdgcn_tensor_load_to_lds(g0, g1, z4, z4, z8, 0);
#else
  __builtin_amdgcn_tensor_load_to_lds(g0, g1, z4, z4, 0);
#endif
}
#endif  // HAVE_TDM

// ---- staging: K chunk (32 x 128, contiguous) + V chunk (128 x 32, strided) -
__device__ __forceinline__ void stage_chunk(__bf16* Kbuf, __bf16* Vbuf,
                                            const __bf16* kg,
                                            const __bf16* vg,
                                            int tid, int wid) {
#if HAVE_TDM
  if (wid == 0)
    tdm_load_2d((unsigned)(size_t)Kbuf, kg, /*tile*/ 128, 32,
                /*tensor*/ 128, 32, /*stride0*/ 128);
  else if (wid == 1)
    tdm_load_2d((unsigned)(size_t)Vbuf, vg, /*tile*/ 32, 128,
                /*tensor*/ 32, 128, /*stride0*/ SEQ_L);
#else
  const int e0 = tid * 16;                       // 4096 elems / 256 thr
  cp16_lds(&Kbuf[e0], kg + e0);
  cp16_lds(&Kbuf[e0 + 8], kg + e0 + 8);
  const int vrow = tid >> 1, vhalf = (tid & 1) * 16;
  cp16_lds(&Vbuf[vrow * 32 + vhalf], vg + (size_t)vrow * SEQ_L + vhalf);
  cp16_lds(&Vbuf[vrow * 32 + vhalf + 8], vg + (size_t)vrow * SEQ_L + vhalf + 8);
#endif
}
__device__ __forceinline__ void stage_wait(int wid) {
#if HAVE_TDM
  if (wid < 2) __builtin_amdgcn_s_wait_tensorcnt(0);
#elif USE_ASYNC_LDS
  asm volatile("s_wait_asynccnt 0x0" ::: "memory");
#endif
}

// ---------------------------------------------------------------------------
// f32 -> bf16 conversion (grid-stride)
// ---------------------------------------------------------------------------
__global__ void f32_to_bf16_kernel(const float* __restrict__ in,
                                   __bf16* __restrict__ out, int n) {
  int i = blockIdx.x * blockDim.x + threadIdx.x;
  int stride = gridDim.x * blockDim.x;
  for (; i < n; i += stride) out[i] = (__bf16)in[i];
}

// ---------------------------------------------------------------------------
// bf16 WMMA GEMM:  C(M,N) = A(M,Kd) @ W(N,Kd)^T , f32 accumulate.
// One wave -> 16x64 tile (4 WMMA tiles, A-fragment reused 4x).
// MODE 0: plain f32; 1: RoPE+scale -> q[b][h][pos][d]; 2: RoPE -> k;
// MODE 3: transposed -> v_t[b][kvh][d][pos].
// ---------------------------------------------------------------------------
template <int MODE>
__global__ __launch_bounds__(256)
void gemm_bf16_wmma(const __bf16* __restrict__ A, const __bf16* __restrict__ W,
                    int Kd, int N,
                    float* __restrict__ outf, __bf16* __restrict__ outb,
                    int heads, float scale) {
  const int lane = threadIdx.x & 31;
  const int wid  = threadIdx.x >> 5;
  const int n0   = (blockIdx.x * 8 + wid) * 64;
  const int m0   = blockIdx.y * 16;

  v8f acc[4];
  #pragma unroll
  for (int j = 0; j < 4; ++j)
    #pragma unroll
    for (int e = 0; e < 8; ++e) acc[j][e] = 0.0f;

  const int am  = lane & 15;
  const int akb = (lane & 16) ? 16 : 0;
  const int bn  = lane & 15;
  const int bkb = (lane & 16) ? 16 : 0;

  const __bf16* arow = A + (size_t)(m0 + am) * Kd + akb;
  const __bf16* brow[4];
  #pragma unroll
  for (int j = 0; j < 4; ++j)
    brow[j] = W + (size_t)(n0 + j * 16 + bn) * Kd + bkb;

  for (int kk = 0; kk < Kd; kk += 32) {
    if (kk + 256 < Kd) {
      __builtin_prefetch(arow + kk + 256, 0, 0);   // global_prefetch_b8
      __builtin_prefetch(brow[0] + kk + 256, 0, 0);
      __builtin_prefetch(brow[2] + kk + 256, 0, 0);
    }
    v8bf alo = *reinterpret_cast<const v8bf*>(arow + kk);
    v8bf ahi = *reinterpret_cast<const v8bf*>(arow + kk + 8);
    v16bf afrag;
    #pragma unroll
    for (int e = 0; e < 8; ++e) { afrag[e] = alo[e]; afrag[8 + e] = ahi[e]; }
    #pragma unroll
    for (int j = 0; j < 4; ++j) {
      v16bf bfrag = *reinterpret_cast<const v16bf*>(brow[j] + kk);
      acc[j] = __builtin_amdgcn_wmma_f32_16x16x32_bf16(
          false, afrag, false, bfrag, (short)0, acc[j], false, false);
    }
  }

  const int colL = lane & 15;
  #pragma unroll
  for (int j = 0; j < 4; ++j) {
    const int f = n0 + j * 16 + colL;
    #pragma unroll
    for (int i = 0; i < 8; ++i) {
      const int r   = ((lane & 16) ? 8 : 0) + i;
      const int row = m0 + r;
      const int b   = row >> 11;          // / SEQ_L
      const int pos = row & (SEQ_L - 1);
      float v = acc[j][i];
      if (MODE == 0) {
        outf[(size_t)row * N + f] = v;
      } else if (MODE == 3) {
        const int h = f >> 7, d = f & 127;
        outb[(((size_t)b * heads + h) * HDK + d) * SEQ_L + pos] = (__bf16)v;
      } else {
        float pv = __shfl_xor(v, 1, 32);   // RoPE partner f^1 in lane^1
        const int d = f & 127;
        float inv_freq = __expf(-(float)(d & ~1) * (LN10000 / 128.0f));
        float sn, cs;
        __sincosf((float)pos * inv_freq, &sn, &cs);
        float o = (f & 1) ? (pv * sn + v * cs) : (v * cs - pv * sn);
        o *= scale;
        const int h = f >> 7;
        outb[(((size_t)b * heads + h) * SEQ_L + pos) * HDK + d] = (__bf16)o;
      }
    }
  }
}

// ---------------------------------------------------------------------------
// Flash-style causal attention, bf16 WMMA, f32 accumulation.
// Block = 8 waves = 128 query rows of one head. K/V chunks (32 wide) are
// DMA'd once per block into double-buffered LDS (TDM tensor_load_to_lds when
// available, per-lane async-to-LDS otherwise) and consumed by all waves;
// the next chunk's DMA overlaps the current chunk's WMMA stream.
// ---------------------------------------------------------------------------
__global__ __launch_bounds__(256)
void attn_wmma(const __bf16* __restrict__ qb, const __bf16* __restrict__ kb_,
               const __bf16* __restrict__ vt, __bf16* __restrict__ ob) {
  __shared__ alignas(128) __bf16 Klds[2][32 * 128];   // [kv][d]
  __shared__ alignas(128) __bf16 Vlds[2][128 * 32];   // [d][kv]
  __shared__ alignas(128) __bf16 Pst[8][16 * 32];     // per-wave P staging

  const int tid  = threadIdx.x;
  const int lane = tid & 31;
  const int wid  = tid >> 5;
  const int h    = blockIdx.y;
  const int b    = blockIdx.z;
  const int qbase = blockIdx.x * 128;
  const int q0    = qbase + wid * 16;
  const int kvh   = h >> 2;              // H/KV = 4
  const int rloc  = (lane & 16) ? 8 : 0;

  // Q fragments: K=128 -> 4 fragments of 32.
  v16bf qf[4];
  {
    const int am  = lane & 15;
    const int akb = (lane & 16) ? 16 : 0;
    const __bf16* qrow =
        qb + (((size_t)b * HEADS + h) * SEQ_L + q0 + am) * HDK + akb;
    #pragma unroll
    for (int kf = 0; kf < 4; ++kf) {
      v8bf lo = *reinterpret_cast<const v8bf*>(qrow + kf * 32);
      v8bf hi = *reinterpret_cast<const v8bf*>(qrow + kf * 32 + 8);
      #pragma unroll
      for (int e = 0; e < 8; ++e) { qf[kf][e] = lo[e]; qf[kf][8 + e] = hi[e]; }
    }
  }

  float mrow[8], lrow[8];
  v8f oacc[8];
  #pragma unroll
  for (int i = 0; i < 8; ++i) { mrow[i] = -1e30f; lrow[i] = 0.0f; }
  #pragma unroll
  for (int j = 0; j < 8; ++j)
    #pragma unroll
    for (int e = 0; e < 8; ++e) oacc[j][e] = 0.0f;

  const int cmaxw = (q0 + 15) >> 5;            // this wave's causal bound
  const int cmaxb = (qbase + 127) >> 5;        // block's causal bound
  const size_t kvplane = ((size_t)b * KVH + kvh);
  const __bf16* kg0 = kb_ + kvplane * SEQ_L * HDK;   // + c*32*128
  const __bf16* vg0 = vt + kvplane * HDK * SEQ_L;    // + c*32 (row stride L)

  // prologue: stage chunk 0 into buffer 0
  stage_chunk(Klds[0], Vlds[0], kg0, vg0, tid, wid);

  for (int c = 0; c <= cmaxb; ++c) {
    stage_wait(wid);
    __syncthreads();                       // chunk c resident & visible
    const int par = c & 1;
    if (c < cmaxb)                         // overlap next DMA with compute
      stage_chunk(Klds[par ^ 1], Vlds[par ^ 1],
                  kg0 + (size_t)(c + 1) * 32 * HDK, vg0 + (c + 1) * 32,
                  tid, wid);

    if (c <= cmaxw) {
      // ---- two 16x16 S subtiles covering KV [32c, 32c+31] ----
      #pragma unroll
      for (int t = 0; t < 2; ++t) {
        const int kv0 = c * 32 + t * 16;
        v8f s;
        #pragma unroll
        for (int e = 0; e < 8; ++e) s[e] = 0.0f;
        const __bf16* krow =
            &Klds[par][(t * 16 + (lane & 15)) * 128 + ((lane & 16) ? 16 : 0)];
        #pragma unroll
        for (int kf = 0; kf < 4; ++kf) {
          v16bf bf = *reinterpret_cast<const v16bf*>(krow + kf * 32);
          s = __builtin_amdgcn_wmma_f32_16x16x32_bf16(
              false, qf[kf], false, bf, (short)0, s, false, false);
        }
        // causal mask + online softmax
        const int colg = kv0 + (lane & 15);
        float tm[8];
        #pragma unroll
        for (int i = 0; i < 8; ++i) {
          const int rowg = q0 + rloc + i;
          float sv = (colg <= rowg) ? s[i] : -1e30f;
          s[i] = sv;
          tm[i] = sv;
        }
        #pragma unroll
        for (int off = 8; off >= 1; off >>= 1)
          #pragma unroll
          for (int i = 0; i < 8; ++i)
            tm[i] = fmaxf(tm[i], __shfl_xor(tm[i], off, 32));
        float p[8], rs[8];
        #pragma unroll
        for (int i = 0; i < 8; ++i) {
          float mn   = fmaxf(mrow[i], tm[i]);
          float corr = __expf(mrow[i] - mn);
          mrow[i] = mn;
          float pv = __expf(s[i] - mn);
          p[i] = pv;
          rs[i] = pv;
          lrow[i] *= corr;
          #pragma unroll
          for (int j = 0; j < 8; ++j) oacc[j][i] *= corr;
        }
        #pragma unroll
        for (int off = 8; off >= 1; off >>= 1)
          #pragma unroll
          for (int i = 0; i < 8; ++i) rs[i] += __shfl_xor(rs[i], off, 32);
        #pragma unroll
        for (int i = 0; i < 8; ++i) lrow[i] += rs[i];
        #pragma unroll
        for (int i = 0; i < 8; ++i)
          Pst[wid][(rloc + i) * 32 + t * 16 + (lane & 15)] = (__bf16)p[i];
      }
      // ---- O += P(16x32) @ V(32x128) ----
      v16bf pf;
      {
        const int am  = lane & 15;
        const int akb = (lane & 16) ? 16 : 0;
        const __bf16* pr = &Pst[wid][am * 32 + akb];
        #pragma unroll
        for (int e = 0; e < 8; ++e) { pf[e] = pr[e]; pf[8 + e] = pr[8 + e]; }
      }
      #pragma unroll
      for (int j = 0; j < 8; ++j) {
        const __bf16* vr =
            &Vlds[par][(j * 16 + (lane & 15)) * 32 + ((lane & 16) ? 16 : 0)];
        v16bf vf = *reinterpret_cast<const v16bf*>(vr);
        oacc[j] = __builtin_amdgcn_wmma_f32_16x16x32_bf16(
            false, pf, false, vf, (short)0, oacc[j], false, false);
      }
    }
  }

  // finalize: divide by row sums, store as (B, L, D) bf16
  #pragma unroll
  for (int i = 0; i < 8; ++i) {
    float inv = 1.0f / lrow[i];
    const int pos = q0 + rloc + i;
    #pragma unroll
    for (int j = 0; j < 8; ++j)
      ob[((size_t)b * SEQ_L + pos) * DIM_D + h * HDK + j * 16 + (lane & 15)] =
          (__bf16)(oacc[j][i] * inv);
  }
}

// ---------------------------------------------------------------------------
// Host-side launch
// ---------------------------------------------------------------------------
extern "C" void kernel_launch(void* const* d_in, const int* in_sizes, int n_in,
                              void* d_out, int out_size, void* d_ws,
                              size_t ws_size, hipStream_t stream) {
  const float* x  = (const float*)d_in[0];
  const float* wq = (const float*)d_in[1];
  const float* wk = (const float*)d_in[2];
  const float* wv = (const float*)d_in[3];
  const float* wo = (const float*)d_in[4];

  const int nX  = 2 * SEQ_L * DIM_D;        // 8,388,608
  const int nWq = DIM_D * DIM_D;            // 4,194,304
  const int nWk = KVH * HDK * DIM_D;        // 1,048,576
  const int nWo = DIM_D * DIM_D;

  char* ws = (char*)d_ws;
  size_t off = 0;
  auto carve = [&](size_t bytes) {
    char* p = ws + off;
    off += (bytes + 255) & ~(size_t)255;
    return p;
  };
  __bf16* x_bf  = (__bf16*)carve((size_t)nX * 2);
  __bf16* wq_bf = (__bf16*)carve((size_t)nWq * 2);
  __bf16* wk_bf = (__bf16*)carve((size_t)nWk * 2);
  __bf16* wv_bf = (__bf16*)carve((size_t)nWk * 2);
  __bf16* wo_bf = (__bf16*)carve((size_t)nWo * 2);
  __bf16* q_bf  = (__bf16*)carve((size_t)nX * 2);                 // (B,H,L,128)
  __bf16* k_bf  = (__bf16*)carve((size_t)2 * KVH * SEQ_L * HDK * 2);
  __bf16* v_t   = (__bf16*)carve((size_t)2 * KVH * HDK * SEQ_L * 2);
  __bf16* o_bf  = (__bf16*)carve((size_t)nX * 2);                 // (B,L,D)

  // 1) convert to bf16
  f32_to_bf16_kernel<<<4096, 256, 0, stream>>>(x, x_bf, nX);
  f32_to_bf16_kernel<<<4096, 256, 0, stream>>>(wq, wq_bf, nWq);
  f32_to_bf16_kernel<<<2048, 256, 0, stream>>>(wk, wk_bf, nWk);
  f32_to_bf16_kernel<<<2048, 256, 0, stream>>>(wv, wv_bf, nWk);
  f32_to_bf16_kernel<<<4096, 256, 0, stream>>>(wo, wo_bf, nWo);

  const int Mtiles = (2 * SEQ_L) / 16;   // 256
  dim3 blk(256);

  // 2) projections (+ fused RoPE / transpose epilogues); block = 512 cols
  gemm_bf16_wmma<1><<<dim3(DIM_D / 512, Mtiles), blk, 0, stream>>>(
      x_bf, wq_bf, DIM_D, DIM_D, nullptr, q_bf, HEADS, QSCALE);
  gemm_bf16_wmma<2><<<dim3((KVH * HDK) / 512, Mtiles), blk, 0, stream>>>(
      x_bf, wk_bf, DIM_D, KVH * HDK, nullptr, k_bf, KVH, 1.0f);
  gemm_bf16_wmma<3><<<dim3((KVH * HDK) / 512, Mtiles), blk, 0, stream>>>(
      x_bf, wv_bf, DIM_D, KVH * HDK, nullptr, v_t, KVH, 1.0f);

  // 3) attention
  attn_wmma<<<dim3(SEQ_L / 128, HEADS, 2), blk, 0, stream>>>(q_bf, k_bf, v_t,
                                                             o_bf);

  // 4) output projection -> f32 result
  gemm_bf16_wmma<0><<<dim3(DIM_D / 512, Mtiles), blk, 0, stream>>>(
      o_bf, wo_bf, DIM_D, DIM_D, (float*)d_out, nullptr, 0, 1.0f);
}
